// DNA_68247030334268
// MI455X (gfx1250) — matrix operand order repaired and checked by reading, not compiled, not run on hardware
//
#include <hip/hip_runtime.h>

#define TT 2048
#define DM 1024
#define VV 32000
#define DFFC 4096
#define NHH 16
#define HDD 64
#define NHOPSC 4
#define CAPC 512

typedef __attribute__((ext_vector_type(16))) __bf16 v16bf;
typedef __attribute__((ext_vector_type(8)))  float  v8f;

union Frag16 { v16bf v; unsigned u[8]; };

__device__ __forceinline__ unsigned short f2bfu(float x) {
  __bf16 b = (__bf16)x;                       // native v_cvt f32->bf16
  return __builtin_bit_cast(unsigned short, b);
}
__device__ __forceinline__ unsigned packbf(float a, float b) {
  return (unsigned)f2bfu(a) | ((unsigned)f2bfu(b) << 16);
}
__device__ __forceinline__ float gelu_f(float x) {
  float x3 = x * x * x;
  return 0.5f * x * (1.0f + tanhf(0.79788456080286535588f * (x + 0.044715f * x3)));
}

// ---------------------------------------------------------------------------
// Generic WMMA GEMM: C[M,N] = epi(A[M,K] @ B) (+R). A row-major f32.
// btrans==0: B is [K,N] row-major.  btrans==1: B is [N,K] row-major (B^T used).
// Block: 128 threads (4 waves), 64x128 tile, wave tile 32x64 (2x4 WMMA frags),
// K step 32, bf16 WMMA f32 accumulate. 12KB LDS.
// ---------------------------------------------------------------------------
__global__ __launch_bounds__(128)
void k_gemm(const float* __restrict__ A, const float* __restrict__ B,
            float* __restrict__ C, int M, int N, int K,
            const float* __restrict__ R, int gelu, int btrans) {
  __shared__ unsigned short As[64 * 32];   // [m][k] bf16
  __shared__ unsigned short Bs[128 * 32];  // [n][k] bf16 (B transposed in LDS)
  int tid  = threadIdx.x;
  int mB   = blockIdx.y * 64, nB = blockIdx.x * 128;
  int lane = tid & 31, wid = tid >> 5;
  int half = lane >> 4, l16 = lane & 15;
  int wm = (wid >> 1) * 32, wn = (wid & 1) * 64;
  v8f acc[2][4] = {};
  for (int k0 = 0; k0 < K; k0 += 32) {
#pragma unroll
    for (int it = 0; it < 8; ++it) {               // A tile: 64x32, 1024 pairs
      int p = tid + 128 * it;
      int row = p >> 4, col = (p & 15) * 2;
      const float* ap = A + (size_t)(mB + row) * K + k0 + col;
      if (it == 0 && k0 + 32 < K) __builtin_prefetch(ap + 32, 0, 0);
      *(unsigned*)&As[row * 32 + col] = packbf(ap[0], ap[1]);
    }
    if (btrans) {
#pragma unroll
      for (int it = 0; it < 16; ++it) {            // 128x32, 2048 pairs
        int p = tid + 128 * it;
        int n = p >> 4, col = (p & 15) * 2;
        const float* bp = B + (size_t)(nB + n) * K + k0 + col;
        if (it == 0 && k0 + 32 < K) __builtin_prefetch(bp + 32, 0, 0);
        *(unsigned*)&Bs[n * 32 + col] = packbf(bp[0], bp[1]);
      }
    } else {
#pragma unroll
      for (int it = 0; it < 32; ++it) {            // 32x128 scalars, coalesced in n
        int idx = tid + 128 * it;
        int kk = idx >> 7, n = idx & 127;
        const float* bp = B + (size_t)(k0 + kk) * N + nB + n;
        if (it == 0 && k0 + 32 < K) __builtin_prefetch(bp + (size_t)32 * N, 0, 0);
        Bs[n * 32 + kk] = f2bfu(*bp);
      }
    }
    __syncthreads();
    Frag16 af[2], bf[4];
#pragma unroll
    for (int i = 0; i < 2; ++i)
#pragma unroll
      for (int j = 0; j < 8; ++j) {                // A layout: k=16*(j>>2)+8*half+2*(j&3)
        int kk = 16 * (j >> 2) + 8 * half + 2 * (j & 3);
        af[i].u[j] = *(const unsigned*)&As[(wm + i * 16 + l16) * 32 + kk];
      }
#pragma unroll
    for (int i = 0; i < 4; ++i)
#pragma unroll
      for (int j = 0; j < 8; ++j) {                // B layout: k=16*(j>>2)+8*half+2*(j&3), n=lane&15
        int kk = 16 * (j >> 2) + 8 * half + 2 * (j & 3);
        bf[i].u[j] = *(const unsigned*)&Bs[(wn + i * 16 + l16) * 32 + kk];
      }
#pragma unroll
    for (int i = 0; i < 2; ++i)
#pragma unroll
      for (int j = 0; j < 4; ++j)
        acc[i][j] = __builtin_amdgcn_wmma_f32_16x16x32_bf16(
            false, af[i].v, false, bf[j].v, (short)0, acc[i][j], false, false);
    __syncthreads();
  }
#pragma unroll
  for (int i = 0; i < 2; ++i)
#pragma unroll
    for (int j = 0; j < 4; ++j) {
      int rbase = mB + wm + i * 16 + 8 * half;     // C layout: m=r+8*half
      int cbase = nB + wn + j * 16 + l16;
#pragma unroll
      for (int r = 0; r < 8; ++r) {
        float vv = acc[i][j][r];
        if (gelu) vv = gelu_f(vv);
        size_t off = (size_t)(rbase + r) * N + cbase;
        if (R) vv += R[off];
        C[off] = vv;
      }
    }
}

// ---------------------------------------------------------------------------
// Flash attention: one wave per (16-query tile, head). Causal, online softmax.
// q/k/v/o are [S, DM] f32 (head h occupies cols h*64..h*64+63), q scaled by `scale` on load.
// ---------------------------------------------------------------------------
__global__ __launch_bounds__(32)
void k_attn(const float* __restrict__ q, const float* __restrict__ k,
            const float* __restrict__ v, float* __restrict__ o, float scale) {
  __shared__ unsigned short Pl[16 * 32];           // probs tile, bf16, [m][key]
  int m0 = blockIdx.x * 16, hh = blockIdx.y;
  int lane = threadIdx.x, half = lane >> 4, l16 = lane & 15;
  const size_t hoff = (size_t)hh * HDD;
  Frag16 qa[2];                                    // Q 16x64 as two A frags
#pragma unroll
  for (int kh = 0; kh < 2; ++kh)
#pragma unroll
    for (int j = 0; j < 8; ++j) {
      int hd = kh * 32 + 16 * (j >> 2) + 8 * half + 2 * (j & 3);
      const float* p = q + (size_t)(m0 + l16) * DM + hoff + hd;
      qa[kh].u[j] = packbf(p[0] * scale, p[1] * scale);
    }
  v8f Oacc[4] = {};
  float mrun[8], lrun[8];
#pragma unroll
  for (int r = 0; r < 8; ++r) { mrun[r] = -1e30f; lrun[r] = 0.f; }
  for (int j0 = 0; j0 < m0 + 16; j0 += 32) {       // 32 keys per step
    v8f sc[2] = {};
#pragma unroll
    for (int nt = 0; nt < 2; ++nt)
#pragma unroll
      for (int kh = 0; kh < 2; ++kh) {
        Frag16 kb;                                 // K^T as B frag (k = hd dim)
#pragma unroll
        for (int j = 0; j < 8; ++j) {
          int key = j0 + nt * 16 + l16;
          int hd = kh * 32 + 16 * (j >> 2) + 8 * half + 2 * (j & 3);
          const float* p = k + (size_t)key * DM + hoff + hd;
          kb.u[j] = packbf(p[0], p[1]);
        }
        sc[nt] = __builtin_amdgcn_wmma_f32_16x16x32_bf16(
            false, qa[kh].v, false, kb.v, (short)0, sc[nt], false, false);
      }
#pragma unroll
    for (int nt = 0; nt < 2; ++nt)
#pragma unroll
      for (int r = 0; r < 8; ++r) {                // causal mask
        int col = j0 + nt * 16 + l16;
        int row = m0 + r + 8 * half;
        if (col > row) sc[nt][r] = -1e9f;
      }
#pragma unroll
    for (int r = 0; r < 8; ++r) {                  // online softmax per row
      float mx = fmaxf(sc[0][r], sc[1][r]);
#pragma unroll
      for (int off = 8; off; off >>= 1) mx = fmaxf(mx, __shfl_xor(mx, off, 32));
      float mn = fmaxf(mrun[r], mx);
      float al = __expf(mrun[r] - mn);
      float p0 = __expf(sc[0][r] - mn);
      float p1 = __expf(sc[1][r] - mn);
      sc[0][r] = p0; sc[1][r] = p1;
      float sm = p0 + p1;
#pragma unroll
      for (int off = 8; off; off >>= 1) sm += __shfl_xor(sm, off, 32);
      lrun[r] = lrun[r] * al + sm;
      mrun[r] = mn;
#pragma unroll
      for (int nt = 0; nt < 4; ++nt) Oacc[nt][r] *= al;
    }
#pragma unroll
    for (int nt = 0; nt < 2; ++nt)                 // stage P to LDS (row-major)
#pragma unroll
      for (int r = 0; r < 8; ++r)
        Pl[(r + 8 * half) * 32 + nt * 16 + l16] = f2bfu(sc[nt][r]);
    asm volatile("s_wait_dscnt 0x0" ::: "memory");
    Frag16 pa;                                     // reload P in A layout
#pragma unroll
    for (int j = 0; j < 8; ++j) {
      int kk = 16 * (j >> 2) + 8 * half + 2 * (j & 3);
      pa.u[j] = *(const unsigned*)&Pl[l16 * 32 + kk];
    }
#pragma unroll
    for (int nt = 0; nt < 4; ++nt) {               // P @ V (k = key index)
      Frag16 vb;
#pragma unroll
      for (int j = 0; j < 8; ++j) {
        int key = j0 + 16 * (j >> 2) + 8 * half + 2 * (j & 3);
        int hd = nt * 16 + l16;
        float f0 = v[(size_t)key * DM + hoff + hd];
        float f1 = v[(size_t)(key + 1) * DM + hoff + hd];
        vb.u[j] = packbf(f0, f1);
      }
      Oacc[nt] = __builtin_amdgcn_wmma_f32_16x16x32_bf16(
          false, pa.v, false, vb.v, (short)0, Oacc[nt], false, false);
    }
  }
#pragma unroll
  for (int nt = 0; nt < 4; ++nt)
#pragma unroll
    for (int r = 0; r < 8; ++r)
      o[(size_t)(m0 + r + 8 * half) * DM + hoff + nt * 16 + l16] =
          Oacc[nt][r] / lrun[r];
}

// ---------------------------------------------------------------------------
__global__ __launch_bounds__(256)
void k_rms(const float* __restrict__ x, const float* __restrict__ ln,
           float* __restrict__ y) {
  __shared__ float red[8];
  __shared__ float rsv;
  int t = blockIdx.x, tid = threadIdx.x;
  float ss = 0.f;
  for (int d = tid; d < DM; d += 256) { float a = x[(size_t)t * DM + d]; ss += a * a; }
#pragma unroll
  for (int off = 16; off; off >>= 1) ss += __shfl_xor(ss, off, 32);
  if ((tid & 31) == 0) red[tid >> 5] = ss;
  __syncthreads();
  if (tid == 0) {
    float tot = 0.f;
    for (int i = 0; i < 8; ++i) tot += red[i];
    rsv = rsqrtf(tot / (float)DM + 1e-6f);
  }
  __syncthreads();
  float rs = rsv;
  for (int d = tid; d < DM; d += 256)
    y[(size_t)t * DM + d] = x[(size_t)t * DM + d] * rs * ln[d];
}

__global__ void k_embed(const int* __restrict__ ids, const float* __restrict__ W,
                        float* __restrict__ h) {
  int t = blockIdx.x; int id = ids[t];
  for (int d = threadIdx.x; d < DM; d += 256)
    h[(size_t)t * DM + d] = W[(size_t)id * DM + d];
}

__global__ void k_ropetab(float* __restrict__ cs, float* __restrict__ sn) {
  int t = blockIdx.x, j = threadIdx.x;             // 32 threads
  float inv = powf(10000.0f, -(float)j / 32.0f);
  float f = (float)t * inv;
  float c = cosf(f), s = sinf(f);
  cs[t * 64 + j] = c; cs[t * 64 + j + 32] = c;
  sn[t * 64 + j] = s; sn[t * 64 + j + 32] = s;
}

__global__ void k_rope(float* __restrict__ q, float* __restrict__ k,
                       const float* __restrict__ cs, const float* __restrict__ sn,
                       int total) {
  int idx = blockIdx.x * 256 + threadIdx.x;
  if (idx >= total) return;
  int d = idx & 31;
  int hh = (idx >> 5) & 15;
  int s = idx >> 9;
  float c1 = cs[s * 64 + d], s1 = sn[s * 64 + d];
  float c2 = cs[s * 64 + d + 32], s2 = sn[s * 64 + d + 32];
  size_t base = (size_t)s * DM + hh * 64;
  float a = q[base + d], b = q[base + d + 32];
  q[base + d] = a * c1 - b * s1;
  q[base + d + 32] = b * c2 + a * s2;
  a = k[base + d]; b = k[base + d + 32];
  k[base + d] = a * c1 - b * s1;
  k[base + d + 32] = b * c2 + a * s2;
}

__global__ __launch_bounds__(256)
void k_router(const float* __restrict__ h, const float* __restrict__ W,
              float* __restrict__ gate, int* __restrict__ tokmask) {
  __shared__ float red[8];
  __shared__ float lg[9];
  int t = blockIdx.x, tid = threadIdx.x;
  for (int e = 0; e < 9; ++e) {
    float ss = 0.f;
    for (int d = tid; d < DM; d += 256) ss += h[(size_t)t * DM + d] * W[(size_t)e * DM + d];
#pragma unroll
    for (int off = 16; off; off >>= 1) ss += __shfl_xor(ss, off, 32);
    if ((tid & 31) == 0) red[tid >> 5] = ss;
    __syncthreads();
    if (tid == 0) { float tot = 0.f; for (int i = 0; i < 8; ++i) tot += red[i]; lg[e] = tot; }
    __syncthreads();
  }
  if (tid == 0) {
    float mx = lg[0];
    for (int e = 1; e < 9; ++e) mx = fmaxf(mx, lg[e]);
    float soft[9]; float sum = 0.f;
    for (int e = 0; e < 9; ++e) { soft[e] = __expf(lg[e] - mx); sum += soft[e]; }
    for (int e = 0; e < 9; ++e) soft[e] /= sum;
    int t1 = 0;
    for (int e = 1; e < 9; ++e) if (lg[e] > lg[t1]) t1 = e;
    int t2 = (t1 == 0) ? 1 : 0;
    for (int e = 0; e < 9; ++e) if (e != t1 && lg[e] > lg[t2]) t2 = e;
    float g[9]; float gs = 0.f;
    for (int e = 0; e < 9; ++e) {
      g[e] = (e == t1 || e == t2) ? ((1.0f - soft[e]) + soft[e]) : 0.f;  // straight-through
      gs += g[e];
    }
    gs = fmaxf(gs, 1e-9f);
    for (int e = 0; e < 9; ++e) gate[t * 9 + e] = g[e] / gs;
    tokmask[t] = (1 << t1) | (1 << t2);
  }
}

__global__ void k_rank(const int* __restrict__ tokmask, int* __restrict__ slot_tok,
                       int* __restrict__ tok_slot) {
  if (threadIdx.x != 0) return;
  int e = blockIdx.x;
  int cnt = 0;
  for (int t = 0; t < TT; ++t) {
    if ((tokmask[t] >> e) & 1) {
      if (cnt < CAPC) { slot_tok[e * CAPC + cnt] = t; tok_slot[e * TT + t] = cnt; }
      else tok_slot[e * TT + t] = -1;
      ++cnt;
    } else tok_slot[e * TT + t] = -1;
  }
  for (int c = cnt; c < CAPC; ++c) slot_tok[e * CAPC + c] = -1;
}

__global__ __launch_bounds__(256)
void k_dispatch(const float* __restrict__ h, const float* __restrict__ gate,
                const float* __restrict__ cs, const float* __restrict__ sn,
                const int* __restrict__ slot_tok, float* __restrict__ xin,
                float* __restrict__ cosr, float* __restrict__ sinr) {
  int c = blockIdx.x, e = blockIdx.y, tid = threadIdx.x;
  int t = slot_tok[e * CAPC + c];
  float g = (t >= 0) ? gate[t * 9 + e] : 0.f;
  size_t ob = ((size_t)e * CAPC + c) * DM;
  for (int d = tid; d < DM; d += 256)
    xin[ob + d] = (t >= 0) ? g * h[(size_t)t * DM + d] : 0.f;
  if ((e & 1) == 0 && tid < 64) {                  // attention experts get rope rows
    int ea = e >> 1;
    size_t cb = ((size_t)ea * CAPC + c) * 64 + tid;
    cosr[cb] = (t >= 0) ? cs[t * 64 + tid] : 0.f;
    sinr[cb] = (t >= 0) ? sn[t * 64 + tid] : 0.f;
  }
}

__global__ __launch_bounds__(256)
void k_combine(float* __restrict__ h, const float* __restrict__ gate,
               const int* __restrict__ tok_slot, const float* __restrict__ eout) {
  int t = blockIdx.x, tid = threadIdx.x;
  float rho = 0.f;
#pragma unroll
  for (int e = 0; e < 8; ++e) rho += gate[t * 9 + e];
  int cslot[8]; float cg[8];
#pragma unroll
  for (int e = 0; e < 8; ++e) { cslot[e] = tok_slot[e * TT + t]; cg[e] = gate[t * 9 + e]; }
  for (int d = tid; d < DM; d += 256) {
    float acc = h[(size_t)t * DM + d] * (1.f - rho);
#pragma unroll
    for (int e = 0; e < 8; ++e) {
      int c = cslot[e];
      if (c >= 0) acc += cg[e] * eout[((size_t)e * CAPC + c) * DM + d];
    }
    h[(size_t)t * DM + d] = acc;
  }
}

// ---------------------------------------------------------------------------
static void gemm(const float* A, const float* B, float* C, int M, int N, int K,
                 const float* R, int gelu, int btrans, hipStream_t s) {
  dim3 g(N / 128, M / 64);
  k_gemm<<<g, 128, 0, s>>>(A, B, C, M, N, K, R, gelu, btrans);
}

// Workspace layout (float offsets)
static const size_t OFF_H    = 0;
static const size_t OFF_XN   = 2097152;
static const size_t OFF_Q    = 4194304;
static const size_t OFF_K    = 6291456;
static const size_t OFF_V    = 8388608;
static const size_t OFF_AO   = 10485760;
static const size_t OFF_MID  = 12582912;   // backbone TT*DFF; experts reuse first 2M
static const size_t OFF_XIN  = 14680064;   // 8*CAP*DM (overlaps dead backbone MID tail)
static const size_t OFF_EOUT = 18874368;   // 8*CAP*DM
static const size_t OFF_COS  = 23068672;
static const size_t OFF_SIN  = 23199744;
static const size_t OFF_COSR = 23330816;
static const size_t OFF_SINR = 23461888;
static const size_t OFF_GATE = 23592960;
static const size_t OFF_INT  = 23613440;

extern "C" void kernel_launch(void* const* d_in, const int* in_sizes, int n_in,
                              void* d_out, int out_size, void* d_ws, size_t ws_size,
                              hipStream_t stream) {
  (void)in_sizes; (void)n_in; (void)out_size; (void)ws_size;
  const int*   ids       = (const int*)d_in[0];
  const float* embW      = (const float*)d_in[1];
  const float* routW     = (const float*)d_in[2];
  const float* bb_aln    = (const float*)d_in[3];
  const float* bb_wq     = (const float*)d_in[4];
  const float* bb_wk     = (const float*)d_in[5];
  const float* bb_wv     = (const float*)d_in[6];
  const float* bb_wo     = (const float*)d_in[7];
  const float* bb_fln    = (const float*)d_in[8];
  const float* bb_w1     = (const float*)d_in[9];
  const float* bb_w2     = (const float*)d_in[10];
  const float* e_aln     = (const float*)d_in[11];
  const float* e_wq      = (const float*)d_in[12];
  const float* e_wk      = (const float*)d_in[13];
  const float* e_wv      = (const float*)d_in[14];
  const float* e_wo      = (const float*)d_in[15];
  const float* e_fln     = (const float*)d_in[16];
  const float* e_w1      = (const float*)d_in[17];
  const float* e_w2      = (const float*)d_in[18];
  const float* lnoutw    = (const float*)d_in[19];
  float* out = (float*)d_out;

  float* ws   = (float*)d_ws;
  float* h    = ws + OFF_H;
  float* xn   = ws + OFF_XN;
  float* q    = ws + OFF_Q;
  float* k    = ws + OFF_K;
  float* v    = ws + OFF_V;
  float* ao   = ws + OFF_AO;
  float* mid  = ws + OFF_MID;
  float* xin  = ws + OFF_XIN;
  float* eout = ws + OFF_EOUT;
  float* cosb = ws + OFF_COS;
  float* sinb = ws + OFF_SIN;
  float* cosr = ws + OFF_COSR;
  float* sinr = ws + OFF_SINR;
  float* gate = ws + OFF_GATE;
  int* slot_tok = (int*)(ws + OFF_INT);
  int* tok_slot = slot_tok + 8 * CAPC;
  int* tokmask  = tok_slot + 8 * TT;

  const float scale = 0.125f;  // HD^-0.5

  k_embed<<<TT, 256, 0, stream>>>(ids, embW, h);
  k_ropetab<<<TT, 32, 0, stream>>>(cosb, sinb);

  // ---- backbone attention ----
  k_rms<<<TT, 256, 0, stream>>>(h, bb_aln, xn);
  gemm(xn, bb_wq, q, TT, DM, DM, nullptr, 0, 0, stream);
  gemm(xn, bb_wk, k, TT, DM, DM, nullptr, 0, 0, stream);
  gemm(xn, bb_wv, v, TT, DM, DM, nullptr, 0, 0, stream);
  k_rope<<<(TT * NHH * 32) / 256, 256, 0, stream>>>(q, k, cosb, sinb, TT * NHH * 32);
  k_attn<<<dim3(TT / 16, NHH), 32, 0, stream>>>(q, k, v, ao, scale);
  gemm(ao, bb_wo, h, TT, DM, DM, /*R=*/h, 0, 0, stream);

  // ---- backbone FFN ----
  k_rms<<<TT, 256, 0, stream>>>(h, bb_fln, xn);
  gemm(xn, bb_w1, mid, TT, DFFC, DM, nullptr, /*gelu=*/1, 0, stream);
  gemm(mid, bb_w2, h, TT, DM, DFFC, /*R=*/h, 0, 0, stream);

  // ---- router hops ----
  for (int hop = 0; hop < NHOPSC; ++hop) {
    k_router<<<TT, 256, 0, stream>>>(h, routW + (size_t)hop * 9 * DM, gate, tokmask);
    k_rank<<<8, 32, 0, stream>>>(tokmask, slot_tok, tok_slot);
    k_dispatch<<<dim3(CAPC, 8), 256, 0, stream>>>(h, gate, cosb, sinb, slot_tok,
                                                  xin, cosr, sinr);
    for (int ea = 0; ea < 4; ++ea) {               // attention experts e = 2*ea
      int e = 2 * ea;
      const float* xe = xin + (size_t)e * CAPC * DM;
      float* ee = eout + (size_t)e * CAPC * DM;
      k_rms<<<CAPC, 256, 0, stream>>>(xe, e_aln + (size_t)ea * DM, xn);
      gemm(xn, e_wq + (size_t)ea * DM * DM, q, CAPC, DM, DM, nullptr, 0, 0, stream);
      gemm(xn, e_wk + (size_t)ea * DM * DM, k, CAPC, DM, DM, nullptr, 0, 0, stream);
      gemm(xn, e_wv + (size_t)ea * DM * DM, v, CAPC, DM, DM, nullptr, 0, 0, stream);
      k_rope<<<(CAPC * NHH * 32) / 256, 256, 0, stream>>>(
          q, k, cosr + (size_t)ea * CAPC * 64, sinr + (size_t)ea * CAPC * 64,
          CAPC * NHH * 32);
      k_attn<<<dim3(CAPC / 16, NHH), 32, 0, stream>>>(q, k, v, ao, scale);
      gemm(ao, e_wo + (size_t)ea * DM * DM, ee, CAPC, DM, DM, nullptr, 0, 0, stream);
    }
    for (int fa = 0; fa < 4; ++fa) {               // FFN experts e = 2*fa+1
      int e = 2 * fa + 1;
      const float* xe = xin + (size_t)e * CAPC * DM;
      float* ee = eout + (size_t)e * CAPC * DM;
      k_rms<<<CAPC, 256, 0, stream>>>(xe, e_fln + (size_t)fa * DM, xn);
      gemm(xn, e_w1 + (size_t)fa * DM * DFFC, mid, CAPC, DFFC, DM, nullptr, 1, 0, stream);
      gemm(mid, e_w2 + (size_t)fa * DFFC * DM, ee, CAPC, DM, DFFC, nullptr, 0, 0, stream);
    }
    k_combine<<<TT, 256, 0, stream>>>(h, gate, tok_slot, eout);
  }

  // ---- final norm + logits ----
  k_rms<<<TT, 256, 0, stream>>>(h, lnoutw, xn);
  gemm(xn, embW, out, TT, VV, DM, nullptr, 0, /*btrans=*/1, stream);
}
